// FeedForwardQuantum_65481071407276
// MI455X (gfx1250) — matrix operand order repaired and checked by reading, not compiled, not run on hardware
//
#include <hip/hip_runtime.h>
#include <stdint.h>

typedef __attribute__((ext_vector_type(16))) __bf16 v16bf;
typedef __attribute__((ext_vector_type(8)))  float  v8f;
typedef __attribute__((ext_vector_type(4)))  float  v4f;
typedef __attribute__((ext_vector_type(2)))  float  v2f;

#define BB 8
#define SS 4096
#define DD 1024
#define QQ 16
#define WAVES_PER_BLOCK 8
#define W2_STRIDE 1040   // padded LDS row stride (floats): kills 2-way bank conflicts

// One wave handles a 16-token tile:
//   GEMM1 (K=1024): v_wmma_f32_16x16x32_bf16, W1 staged in LDS in B-fragment order
//   epilogue: relu + native cos, transposed through LDS scratch into A-fragments
//   GEMM2 (K=16):  v_wmma_f32_16x16x4_f32 over 64 column tiles, W2 served from LDS
// W2 is staged per-block with CDNA5 async global->LDS DMA (ASYNCcnt).
__global__ __launch_bounds__(256) void ffq_fused(
    const float* __restrict__ x, const float* __restrict__ W1,
    const float* __restrict__ b1, const float* __restrict__ theta,
    const float* __restrict__ W2, const float* __restrict__ b2v,
    float* __restrict__ out)
{
    // W1 as bf16, pre-swizzled into this kernel's contiguous-k B-fragment order (same as A):
    //   chunk c (32 k's), lane l, elem i -> W1[(c*32 + (l>>4)*16 + i)*16 + (l&15)]
    __shared__ __align__(32) __bf16 sW1[(DD / 32) * 512];     // 32 KB
    __shared__ __align__(16) float  sW2[QQ * W2_STRIDE];      // 66.5 KB, padded rows
    __shared__ float sQ[WAVES_PER_BLOCK][16 * 20];            // 10 KB transpose scratch

    const int tid = threadIdx.x;

    // ---- stage W2 into LDS via async global->LDS DMA (no VGPR round trip) ----
    {
        const uint32_t lds0 = (uint32_t)(uintptr_t)&sW2[0];
        const uint64_t g0   = (uint64_t)(uintptr_t)W2;
#pragma unroll
        for (int k = 0; k < QQ; ++k) {                 // one 4 KB row per pass
            uint32_t ldsa = lds0 + (uint32_t)(k * (W2_STRIDE * 4)) + tid * 16;
            uint64_t ga   = g0 + (uint64_t)(k * (DD * 4)) + (uint64_t)(tid * 16);
            asm volatile("global_load_async_to_lds_b128 %0, %1, off"
                         :: "v"(ldsa), "v"(ga) : "memory");
        }
    }

    // ---- stage W1 into LDS (needs f32->bf16 convert, so via VGPRs) ----
    for (int s = tid; s < DD * QQ; s += 256) {
        const int k = s >> 4, q = s & 15;
        const int c = k >> 5, kr = k & 31;
        sW1[c * 512 + ((((kr >> 4) << 4) | q) << 4) + (kr & 15)] = (__bf16)W1[s];
    }
    asm volatile("s_wait_asynccnt 0x0" ::: "memory");  // drain this wave's async DMA
    __syncthreads();

    const int lane   = tid & 31;
    const int wave   = tid >> 5;
    const int tile   = blockIdx.x * WAVES_PER_BLOCK + wave;  // 16-token tile
    const int token0 = tile << 4;
    const int mrow   = lane & 15;   // A row / C column owner
    const int half   = lane >> 4;   // 0 or 1
    const int ncol   = mrow;        // B/C column index for this lane

    // ---------------- GEMM1: h = x @ W1 (bf16 WMMA, K chunks of 32) ----------------
    const float* xrow = x + (size_t)(token0 + mrow) * DD;
    const int a_off = half * 16;    // lanes 16-31 hold K+16..K+31
    v8f c1 = {};

    for (int k0 = 0; k0 < DD; k0 += 32) {
        if (k0 + 32 < DD)
            __builtin_prefetch(xrow + k0 + 32 + a_off, 0, 3);  // global_prefetch_b8
        // A fragment: 16 bf16 per lane (K = k0+a_off+0..15)
        v4f f0 = *(const v4f*)(xrow + k0 + a_off);
        v4f f1 = *(const v4f*)(xrow + k0 + a_off + 4);
        v4f f2 = *(const v4f*)(xrow + k0 + a_off + 8);
        v4f f3 = *(const v4f*)(xrow + k0 + a_off + 12);
        v16bf a;
#pragma unroll
        for (int i = 0; i < 4; ++i) {
            a[i]      = (__bf16)f0[i];
            a[4 + i]  = (__bf16)f1[i];
            a[8 + i]  = (__bf16)f2[i];
            a[12 + i] = (__bf16)f3[i];
        }
        // B fragment: one contiguous 32B LDS read per lane
        v16bf bfrag = *(const v16bf*)&sW1[(k0 >> 5) * 512 + lane * 16];
        c1 = __builtin_amdgcn_wmma_f32_16x16x32_bf16(false, a, false, bfrag,
                                                     (short)0, c1, false, false);
    }

    // ------- epilogue: relu + native cos; write q tile to LDS (row m, col q) -------
    {
        const float bb = b1[ncol];
        const float th = theta[ncol];
#pragma unroll
        for (int r = 0; r < 8; ++r) {
            const int m = r + (half << 3);
            float h = c1[r] + bb;
            h = fmaxf(h, 0.0f);
            sQ[wave][m * 20 + ncol] = __cosf(h + th);   // v_cos_f32 path
        }
    }
    __builtin_amdgcn_wave_barrier();   // wave-private tile: just block reordering

    // A2 fragments for f32 16x16x4: lane holds A(m, 2*half + {0,1}) per K-step
    v2f a2[4];
    {
        const float* rowp = &sQ[wave][mrow * 20];
#pragma unroll
        for (int kk = 0; kk < 4; ++kk) {
            const int kb = kk * 4 + half * 2;
            a2[kk].x = rowp[kb];
            a2[kk].y = rowp[kb + 1];
        }
    }

    // ---------------- GEMM2: out = q @ W2 + b2 (f32 WMMA, 64 column tiles) ----------------
    float* orow = out + (size_t)token0 * DD;
    for (int nt = 0; nt < 64; ++nt) {
        const int col = (nt << 4) | ncol;
        v8f c2 = {};
#pragma unroll
        for (int kk = 0; kk < 4; ++kk) {
            const int kr = kk * 4 + half * 2;
            v2f wb;                       // B(k, n): V0 = K=2*half, V1 = K=2*half+1
            wb.x = sW2[kr * W2_STRIDE + col];
            wb.y = sW2[(kr + 1) * W2_STRIDE + col];
            c2 = __builtin_amdgcn_wmma_f32_16x16x4_f32(false, a2[kk], false, wb,
                                                       (short)0, c2, false, false);
        }
        const float bias = b2v[col];
#pragma unroll
        for (int r = 0; r < 8; ++r) {
            const int m = r + (half << 3);
            orow[(size_t)m * DD + col] = c2[r] + bias;   // 64B-coalesced per row
        }
    }
}

extern "C" void kernel_launch(void* const* d_in, const int* in_sizes, int n_in,
                              void* d_out, int out_size, void* d_ws, size_t ws_size,
                              hipStream_t stream) {
    const float* x     = (const float*)d_in[0];
    const float* W1    = (const float*)d_in[1];
    const float* b1    = (const float*)d_in[2];
    const float* theta = (const float*)d_in[3];
    const float* W2    = (const float*)d_in[4];
    const float* b2    = (const float*)d_in[5];
    float* out = (float*)d_out;

    const int tiles = (BB * SS) / 16;                 // 2048 token tiles
    dim3 grid(tiles / WAVES_PER_BLOCK);               // 256 blocks
    dim3 block(256);                                  // 8 waves/block (wave32)
    ffq_fused<<<grid, block, 0, stream>>>(x, W1, b1, theta, W2, b2, out);
}
